// ODEFunc_27590869910220
// MI455X (gfx1250) — matrix operand order repaired and checked by reading, not compiled, not run on hardware
//
#include <hip/hip_runtime.h>
#include <hip/hip_bf16.h>

typedef __attribute__((ext_vector_type(16))) _Float16 v16h;
typedef __attribute__((ext_vector_type(8)))  _Float16 v8h;
typedef __attribute__((ext_vector_type(8)))  float    v8f;

#define BN   262144
#define HN   192

// Packed-weight layout in d_ws / LDS (units: halves). Strides padded so that
// 16 lanes doing ds_load_b128 hit distinct banks (dword stride %64 spread).
#define W1T_OFF     0
#define W1T_STRIDE  168            // K padded 145->160, stride 168 (336B, 16B aligned)
#define W1T_SIZE    (192*168)      // 32256
#define W2T_OFF     W1T_SIZE
#define W2T_STRIDE  200            // 400B rows, 16B aligned, conflict-free
#define W2T_SIZE    (192*200)      // 38400
#define W3T_OFF     (W2T_OFF + W2T_SIZE)   // 70656
#define W3T_STRIDE  200
#define W3T_SIZE    (16*200)       // 3200
#define WPACK_HALVES (W3T_OFF + W3T_SIZE)  // 73856  (147712 bytes in d_ws)

#define ACT_OFF      WPACK_HALVES
#define ACT_STRIDE   200
#define ACT_PER_BUF  (16*ACT_STRIDE)       // 3200
#define ACT_PER_WAVE (2*ACT_PER_BUF)       // 6400 (primal + tangent)
#define WAVES        8
#define LDS_HALVES   (ACT_OFF + WAVES*ACT_PER_WAVE)   // 125056
#define LDS_BYTES    (LDS_HALVES*2)                   // 250112 (< 320KB WGP LDS)

// ---------------- helpers ----------------

__device__ __forceinline__ v8f zero8f() {
  v8f z;
#pragma unroll
  for (int i = 0; i < 8; ++i) z[i] = 0.0f;
  return z;
}

__device__ __forceinline__ v8h zero8h() {
  v8h z;
#pragma unroll
  for (int i = 0; i < 8; ++i) z[i] = (_Float16)0.0f;
  return z;
}

__device__ __forceinline__ v16h combine16(v8h lo, v8h hi) {
  v16h r;
#pragma unroll
  for (int i = 0; i < 8; ++i) { r[i] = lo[i]; r[8 + i] = hi[i]; }
  return r;
}

__device__ __forceinline__ v8f wmma16(v16h a, v16h b, v8f c) {
  // D = A(16x32 f16) x B(32x16 f16) + C(16x16 f32)
  return __builtin_amdgcn_wmma_f32_16x16x32_f16(
      /*neg_a=*/false, a, /*neg_b=*/false, b,
      /*c_mod=*/(short)0, c, /*reuse_a=*/false, /*reuse_b=*/false);
}

// B fragment (32x16) from LDS weight tile stored N-major: Wt[n][k].
// lane<16: N=lane, lo=K(kk*32+0..7), hi=K(kk*32+16..23) ; lane>=16: N=lane-16, lo=K(kk*32+8..15), hi=K(kk*32+24..31).
__device__ __forceinline__ v16h ldsBfrag(const _Float16* base, int stride,
                                         int kk, int nn, int hid, int lr) {
  const int n = nn * 16 + lr;
  const int k = kk * 32 + hid * 8;
  const v8h* p = (const v8h*)(base + n * stride + k);
  return combine16(p[0], p[2]);
}

// A fragment (16x32) from an LDS activation buffer (row-major, stride 200).
// lane<16: M=lane, lo=K(kk*32+0..7), hi=K(kk*32+16..23)
// lane>=16: M=lane-16, lo=K(kk*32+8..15), hi=K(kk*32+24..31)
__device__ __forceinline__ v16h ldsAfrag(const _Float16* buf, int kk, int hid, int lr) {
  const int klo = kk * 32 + hid * 8;
  const _Float16* row = buf + lr * ACT_STRIDE;
  v8h lo = *(const v8h*)(row + klo);
  v8h hi = *(const v8h*)(row + klo + 16);
  return combine16(lo, hi);
}

// 8 contiguous input features [kb, kb+8) of the concatenated vector
// x = [y(16) | h(128) | t(1) | pad->160], converted to f16.
__device__ __forceinline__ v8h loadX8(const float* __restrict__ y,
                                      const float* __restrict__ h,
                                      float tval, int row, int kb) {
  float f[8];
  if (kb < 16) {
    const float4* p = (const float4*)(y + (size_t)row * 16 + kb);
    float4 u = p[0], v = p[1];
    f[0]=u.x; f[1]=u.y; f[2]=u.z; f[3]=u.w; f[4]=v.x; f[5]=v.y; f[6]=v.z; f[7]=v.w;
  } else if (kb < 144) {
    const float4* p = (const float4*)(h + (size_t)row * 128 + (kb - 16));
    float4 u = p[0], v = p[1];
    f[0]=u.x; f[1]=u.y; f[2]=u.z; f[3]=u.w; f[4]=v.x; f[5]=v.y; f[6]=v.z; f[7]=v.w;
  } else if (kb == 144) {
    f[0] = tval;
#pragma unroll
    for (int i = 1; i < 8; ++i) f[i] = 0.0f;
  } else {
#pragma unroll
    for (int i = 0; i < 8; ++i) f[i] = 0.0f;
  }
  v8h r;
#pragma unroll
  for (int i = 0; i < 8; ++i) r[i] = (_Float16)f[i];
  return r;
}

// exact-GELU value + derivative, branch-free.
// erf via Abramowitz-Stegun 7.1.26 (|err| < 1.5e-7); its exp(-u^2) with
// u = x/sqrt(2) equals exp(-x^2/2), shared with the pdf term of gelu'.
__device__ __forceinline__ void gelu_both(float x, float& g, float& dg) {
  const float kInvSqrt2 = 0.7071067811865476f;
  const float kInvSqrt2Pi = 0.3989422804014327f;
  float u = x * kInvSqrt2;
  float au = fabsf(u);
  float t = __builtin_amdgcn_rcpf(1.0f + 0.3275911f * au);
  float p = t * (0.254829592f +
           t * (-0.284496736f +
           t * (1.421413741f +
           t * (-1.453152027f +
           t * 1.061405429f))));
  float e = __expf(-au * au);          // = exp(-x^2/2)
  float erf_abs = 1.0f - p * e;
  float er = copysignf(erf_abs, u);
  float cdf = 0.5f * (1.0f + er);
  g = x * cdf;
  dg = cdf + x * kInvSqrt2Pi * e;
}

// ---------------- prep: pack weights f32 -> f16, transposed + padded ----------------

__global__ void prep_weights(const float* __restrict__ W1,
                             const float* __restrict__ W2,
                             const float* __restrict__ W3,
                             _Float16* __restrict__ wp) {
  int i = blockIdx.x * 256 + threadIdx.x;
  if (i >= WPACK_HALVES) return;
  float v = 0.0f;
  if (i < W1T_SIZE) {
    int n = i / W1T_STRIDE, k = i % W1T_STRIDE;
    if (k < 145) v = W1[(size_t)k * 192 + n];
  } else if (i < W3T_OFF) {
    int j = i - W2T_OFF;
    int n = j / W2T_STRIDE, k = j % W2T_STRIDE;
    if (k < 192) v = W2[(size_t)k * 192 + n];
  } else {
    int j = i - W3T_OFF;
    int n = j / W3T_STRIDE, k = j % W3T_STRIDE;
    if (k < 192) v = W3[(size_t)k * 16 + n];
  }
  wp[i] = (_Float16)v;
}

// ---------------- main: fused MLP + JVP ----------------

__global__ __launch_bounds__(256, 1) void ode_main(
    const float* __restrict__ tp, const float* __restrict__ y,
    const float* __restrict__ hin, const float* __restrict__ eps,
    const float* __restrict__ b1, const float* __restrict__ b2,
    const float* __restrict__ b3, const _Float16* __restrict__ wp,
    float* __restrict__ out_f, float* __restrict__ out_dlogp,
    float* __restrict__ out_dh) {
  extern __shared__ _Float16 lds[];
  const int tid = threadIdx.x;

  // cooperative stage of packed weights into LDS (b128 copies)
  {
    v8h* d = (v8h*)lds;
    const v8h* s = (const v8h*)wp;
    for (int i = tid; i < WPACK_HALVES / 8; i += 256) d[i] = s[i];
  }
  __syncthreads();

  const int lane = tid & 31;
  const int wave = tid >> 5;
  const int hid = lane >> 4;    // which 16-lane half
  const int lr = lane & 15;     // lane index within half
  const int tile = blockIdx.x * WAVES + wave;
  const int r0 = tile * 16;
  const float tval = tp[0];
  const int myrow = r0 + lr;

  float b1v[12], b2v[12];
#pragma unroll
  for (int nn = 0; nn < 12; ++nn) {
    b1v[nn] = b1[nn * 16 + lr];
    b2v[nn] = b2[nn * 16 + lr];
  }
  const float b3v = b3[lr];

  _Float16* aP = lds + ACT_OFF + wave * ACT_PER_WAVE;  // primal activations
  _Float16* aT = aP + ACT_PER_BUF;                     // tangent activations

  // ---------- layer 1: z1 = X @ W1, z1t = [eps|0] @ W1 ----------
  // A fragments hoisted into registers; nn-outer so only one primal+tangent
  // accumulator pair is live at a time (low VGPR pressure -> deep ds pipelining).
  v16h AX[5];
#pragma unroll
  for (int kk = 0; kk < 5; ++kk) {   // K = 160 padded
    const int klo = kk * 32 + hid * 8;
    AX[kk] = combine16(loadX8(y, hin, tval, myrow, klo),
                       loadX8(y, hin, tval, myrow, klo + 16));
  }
  v16h AE;
  {
    const float4* pe = (const float4*)(eps + (size_t)myrow * 16 + hid * 8);
    float4 u = pe[0], v = pe[1];
    v8h lo;
    lo[0]=(_Float16)u.x; lo[1]=(_Float16)u.y; lo[2]=(_Float16)u.z; lo[3]=(_Float16)u.w;
    lo[4]=(_Float16)v.x; lo[5]=(_Float16)v.y; lo[6]=(_Float16)v.z; lo[7]=(_Float16)v.w;
    AE = combine16(lo, zero8h());   // K 16..31 rows of W1 see zeros
  }

#pragma unroll
  for (int nn = 0; nn < 12; ++nn) {
    v16h B0 = ldsBfrag(lds + W1T_OFF, W1T_STRIDE, 0, nn, hid, lr);
    v8f t = wmma16(AE, B0, zero8f());      // tangent: only kk=0 contributes
    v8f p = wmma16(AX[0], B0, zero8f());
#pragma unroll
    for (int kk = 1; kk < 5; ++kk) {
      v16h Bf = ldsBfrag(lds + W1T_OFF, W1T_STRIDE, kk, nn, hid, lr);
      p = wmma16(AX[kk], Bf, p);
    }
    // GELU(z1+b1), gelu'(z1+b1)*z1t  ->  LDS f16
#pragma unroll
    for (int r = 0; r < 8; ++r) {
      const int M = r + hid * 8;
      float z = p[r] + b1v[nn];
      float g, dg;
      gelu_both(z, g, dg);
      aP[M * ACT_STRIDE + nn * 16 + lr] = (_Float16)g;
      aT[M * ACT_STRIDE + nn * 16 + lr] = (_Float16)(dg * t[r]);
    }
  }

  // ---------- layer 2: z2 = a1 @ W2 (primal + tangent share B fragments) ----------
  v16h Ap2[6], At2[6];
#pragma unroll
  for (int kk = 0; kk < 6; ++kk) {   // K = 192
    Ap2[kk] = ldsAfrag(aP, kk, hid, lr);
    At2[kk] = ldsAfrag(aT, kk, hid, lr);
  }
#pragma unroll
  for (int nn = 0; nn < 12; ++nn) {
    v8f p = zero8f(), t = zero8f();
#pragma unroll
    for (int kk = 0; kk < 6; ++kk) {
      v16h Bf = ldsBfrag(lds + W2T_OFF, W2T_STRIDE, kk, nn, hid, lr);
      p = wmma16(Ap2[kk], Bf, p);
      t = wmma16(At2[kk], Bf, t);
    }
#pragma unroll
    for (int r = 0; r < 8; ++r) {
      const int M = r + hid * 8;
      float z = p[r] + b2v[nn];
      float g, dg;
      gelu_both(z, g, dg);
      // safe overwrite: in-wave LDS ordering puts these stores after the
      // Ap2/At2 loads above
      aP[M * ACT_STRIDE + nn * 16 + lr] = (_Float16)g;
      aT[M * ACT_STRIDE + nn * 16 + lr] = (_Float16)(dg * t[r]);
    }
  }

  // ---------- layer 3: f = a2 @ W3 + b3 ; Jeps = a2t @ W3 ----------
  v16h Ap3[6], At3[6];
#pragma unroll
  for (int kk = 0; kk < 6; ++kk) {
    Ap3[kk] = ldsAfrag(aP, kk, hid, lr);
    At3[kk] = ldsAfrag(aT, kk, hid, lr);
  }
  v8f f3 = zero8f(), t3 = zero8f();
#pragma unroll
  for (int kk = 0; kk < 6; ++kk) {
    v16h Bf = ldsBfrag(lds + W3T_OFF, W3T_STRIDE, kk, 0, hid, lr);
    f3 = wmma16(Ap3[kk], Bf, f3);
    t3 = wmma16(At3[kk], Bf, t3);
  }

  // outputs: f, dlogp = -sum_n(Jeps * eps)
#pragma unroll
  for (int r = 0; r < 8; ++r) {
    const int M = r + hid * 8;
    const int row = r0 + M;
    out_f[(size_t)row * 16 + lr] = f3[r] + b3v;
    float v = t3[r] * eps[(size_t)row * 16 + lr];
    v += __shfl_xor(v, 1);
    v += __shfl_xor(v, 2);
    v += __shfl_xor(v, 4);
    v += __shfl_xor(v, 8);
    if (lr == 0) out_dlogp[row] = -v;
  }

  // dh = 0 for this wave's 16 rows (16*128 floats = 512 float4)
  float4 z4 = make_float4(0.0f, 0.0f, 0.0f, 0.0f);
  float4* dhp = (float4*)(out_dh + (size_t)r0 * 128);
#pragma unroll 4
  for (int i = lane; i < 512; i += 32) dhp[i] = z4;
}

// ---------------- launch ----------------

extern "C" void kernel_launch(void* const* d_in, const int* in_sizes, int n_in,
                              void* d_out, int out_size, void* d_ws, size_t ws_size,
                              hipStream_t stream) {
  (void)in_sizes; (void)n_in; (void)out_size; (void)ws_size;
  const float* tp  = (const float*)d_in[0];
  const float* y   = (const float*)d_in[1];
  // d_in[2] = logp (unused by the outputs)
  const float* hin = (const float*)d_in[3];
  const float* eps = (const float*)d_in[4];
  const float* W1  = (const float*)d_in[5];
  const float* b1  = (const float*)d_in[6];
  const float* W2  = (const float*)d_in[7];
  const float* b2  = (const float*)d_in[8];
  const float* W3  = (const float*)d_in[9];
  const float* b3  = (const float*)d_in[10];

  float* out_f = (float*)d_out;
  float* out_dlogp = out_f + (size_t)BN * 16;
  float* out_dh = out_dlogp + BN;
  _Float16* wp = (_Float16*)d_ws;

  hipFuncSetAttribute(reinterpret_cast<const void*>(&ode_main),
                      hipFuncAttributeMaxDynamicSharedMemorySize, LDS_BYTES);

  prep_weights<<<(WPACK_HALVES + 255) / 256, 256, 0, stream>>>(W1, W2, W3, wp);
  // 2048 blocks * 8 waves * 16 rows = 262144 rows
  ode_main<<<BN / 128, 256, LDS_BYTES, stream>>>(tp, y, hin, eps, b1, b2, b3, wp,
                                                 out_f, out_dlogp, out_dh);
}
